// DenseGATLayer_52518860096139
// MI455X (gfx1250) — matrix operand order repaired and checked by reading, not compiled, not run on hardware
//
#include <hip/hip_runtime.h>

#define N_NODES 4096
#define IN_DIM  256
#define HIDDEN  64
#define HEADS   8
#define HH      (HEADS * HIDDEN)   // 512
#define NEG_SLOPE 0.2f
#define MASK_FILL -10000.0f

typedef _Float16 half16 __attribute__((ext_vector_type(16)));
typedef float    f32x8  __attribute__((ext_vector_type(8)));
typedef float    f32x4v __attribute__((ext_vector_type(4)));
typedef float    f32x2  __attribute__((ext_vector_type(2)));

// ---------------------------------------------------------------------------
// Kernel 1: h = x @ W^T  via V_WMMA_F32_16X16X4_F32 (full fp32 precision).
// One wave per 16x16 output tile. Also emits hT16[col][n] (f16, transposed)
// so the attention kernel's B operands are contiguous along the j dimension.
// ---------------------------------------------------------------------------
__global__ __launch_bounds__(256) void gemm_h_kernel(
    const float* __restrict__ x, const float* __restrict__ W,
    float* __restrict__ h, _Float16* __restrict__ hT16) {
  const int lane = threadIdx.x & 31;
  const int wave = threadIdx.x >> 5;
  const int tile = blockIdx.x * 8 + wave;       // 8192 tiles total
  const int mt = tile >> 5;                     // 256 row tiles
  const int nt = tile & 31;                     // 32  col tiles
  const int m0 = mt * 16, n0 = nt * 16;
  const int m16 = lane & 15;
  const int half_id = lane >> 4;

  const float* arow = x + (size_t)(m0 + m16) * IN_DIM;   // A row (M = m16)
  const float* brow = W + (size_t)(n0 + m16) * IN_DIM;   // B col n = W row n

  f32x8 acc = {};
#pragma unroll 4
  for (int k = 0; k < IN_DIM; k += 4) {
    const int kb = k + half_id * 2;             // lanes 0-15: K k,k+1 ; 16-31: K k+2,k+3
    f32x2 a = *(const f32x2*)(arow + kb);
    f32x2 b = *(const f32x2*)(brow + kb);
    acc = __builtin_amdgcn_wmma_f32_16x16x4_f32(
        false, a, false, b, (short)0, acc, false, false);
  }

  const int n = n0 + m16;                       // C layout: N = lane&15
#pragma unroll
  for (int r = 0; r < 8; ++r) {                 // rows r (+8 for upper half)
    const int row = m0 + r + 8 * half_id;
    const float v = acc[r];
    h[(size_t)row * HH + n] = v;
    hT16[(size_t)n * N_NODES + row] = (_Float16)v;
  }
}

// ---------------------------------------------------------------------------
// Kernel 2: per-(node, head) attention scores, stored transposed [head][node]
// so kernel 3 reads contiguous 32-element chunks along j.
// ---------------------------------------------------------------------------
__global__ __launch_bounds__(256) void scores_kernel(
    const float* __restrict__ h, const float* __restrict__ attn_src,
    const float* __restrict__ attn_dst,
    float* __restrict__ srcT, float* __restrict__ dstT) {
  const int t = blockIdx.x * 256 + threadIdx.x; // 32768 threads
  const int n = t >> 3;
  const int hd = t & 7;
  const float* hp = h + (size_t)n * HH + hd * HIDDEN;
  const float* as = attn_src + hd * HIDDEN;
  const float* ad = attn_dst + hd * HIDDEN;
  float s = 0.0f, d = 0.0f;
#pragma unroll 8
  for (int i = 0; i < HIDDEN; ++i) {
    const float v = hp[i];
    s += v * as[i];
    d += v * ad[i];
  }
  srcT[(size_t)hd * N_NODES + n] = s;
  dstT[(size_t)hd * N_NODES + n] = d;
}

// ---------------------------------------------------------------------------
// Kernel 3: fused masked-softmax attention (flash style).
// Block = 8 waves = 8 heads for one 16-row query block. Each wave streams j in
// chunks of 32, builds the 16x32 P tile directly in the f16 A-matrix register
// layout, runs an online softmax (cross-lane pair reductions via shfl_xor 16),
// and accumulates P @ h_block with v_wmma_f32_16x16x32_f16. The accumulator
// rescale (shuffle broadcast + 32 mults) is skipped via a wave-uniform branch
// whenever no lane's running max changed — the common case at 5% edge density.
// Heads are averaged through LDS; output written deterministically.
// ---------------------------------------------------------------------------
__global__ __launch_bounds__(256) void gat_attn_kernel(
    const unsigned char* __restrict__ adj,
    const float* __restrict__ srcT, const float* __restrict__ dstT,
    const _Float16* __restrict__ hT16, float* __restrict__ out) {
  __shared__ float ldsO[HEADS * 16 * HIDDEN];   // 32 KB

  const int lane = threadIdx.x & 31;
  const int head = threadIdx.x >> 5;
  const int i0 = blockIdx.x * 16;
  const int m16 = lane & 15;
  const int half_id = lane >> 4;
  const int b = half_id * 8;                    // K-run base for A layout

  const float srcv = srcT[(size_t)head * N_NODES + i0 + m16];
  const unsigned char* adjrow = adj + (size_t)(i0 + m16) * N_NODES;
  const float* dstrow = dstT + (size_t)head * N_NODES;
  const _Float16* bbase = hT16 + (size_t)head * HIDDEN * N_NODES;

  f32x8 acc0 = {}, acc1 = {}, acc2 = {}, acc3 = {};
  float mrun = -1.0e30f;
  float lrun = 0.0f;

  for (int j0 = 0; j0 < N_NODES; j0 += 32) {
    // --- adjacency bytes for this lane's 16 K positions (two 8-byte runs) ---
    const uint2 ad1 = *(const uint2*)(adjrow + j0 + b);
    const uint2 ad2 = *(const uint2*)(adjrow + j0 + 16 + b);
    const unsigned aw[4] = {ad1.x, ad1.y, ad2.x, ad2.y};

    // --- dst scores: same two 8-float runs ---
    const f32x4v d0 = *(const f32x4v*)(dstrow + j0 + b);
    const f32x4v d1 = *(const f32x4v*)(dstrow + j0 + b + 4);
    const f32x4v d2 = *(const f32x4v*)(dstrow + j0 + 16 + b);
    const f32x4v d3 = *(const f32x4v*)(dstrow + j0 + 16 + b + 4);
    float dv[16];
#pragma unroll
    for (int q = 0; q < 4; ++q) { dv[q] = d0[q]; dv[4 + q] = d1[q];
                                  dv[8 + q] = d2[q]; dv[12 + q] = d3[q]; }

    // --- logits: leaky_relu(src+dst), then mask ---
    float sv[16];
    float mloc = -1.0e30f;
#pragma unroll
    for (int kk = 0; kk < 16; ++kk) {
      float t = srcv + dv[kk];
      t = (t > 0.0f) ? t : NEG_SLOPE * t;
      const unsigned byte = (aw[kk >> 2] >> ((kk & 3) * 8)) & 0xFFu;
      if (byte == 0u) t = MASK_FILL;
      sv[kk] = t;
      mloc = fmaxf(mloc, t);
    }

    // --- online softmax update (row split across lane pair L / L+16) ---
    const float mpair = fmaxf(mloc, __shfl_xor(mloc, 16, 32));
    const float mnew = fmaxf(mrun, mpair);

    // Rescale accumulators only if some lane's running max actually changed
    // (wave-uniform branch -> s_cbranch; skips 8 bpermutes + 32 mults).
    if (__any(mnew > mrun)) {
      const float scale = __expf(mrun - mnew);
      float sc[8];
#pragma unroll
      for (int r = 0; r < 8; ++r) sc[r] = __shfl(scale, b + r, 32);
#pragma unroll
      for (int r = 0; r < 8; ++r) {
        acc0[r] *= sc[r]; acc1[r] *= sc[r]; acc2[r] *= sc[r]; acc3[r] *= sc[r];
      }
      lrun *= scale;
    }
    mrun = mnew;

    union { half16 v; _Float16 e[16]; } A;
    float lsum = 0.0f;
#pragma unroll
    for (int kk = 0; kk < 16; ++kk) {
      const float p = __expf(sv[kk] - mnew);
      lsum += p;
      A.e[kk] = (_Float16)p;
    }
    lrun += lsum + __shfl_xor(lsum, 16, 32);

    // --- P @ h_block : 4 WMMAs covering 64 hidden cols ---
    const int joff = j0 + half_id * 8;          // B layout: two 8-K runs (b, 16+b), matching A
#pragma unroll
    for (int c = 0; c < 4; ++c) {
      const _Float16* bp = bbase + (size_t)(c * 16 + m16) * N_NODES + joff;
      union { half16 v; uint4 q[2]; } B;
      B.q[0] = *(const uint4*)(bp);
      B.q[1] = *(const uint4*)(bp + 16);
      f32x8* accp = (c == 0) ? &acc0 : (c == 1) ? &acc1 : (c == 2) ? &acc2 : &acc3;
      *accp = __builtin_amdgcn_wmma_f32_16x16x32_f16(
          false, A.v, false, B.v, (short)0, *accp, false, false);
    }
  }

  // --- finalize: divide by row sums, average heads through LDS ---
  float linv[8];
#pragma unroll
  for (int r = 0; r < 8; ++r)
    linv[r] = 0.125f / __shfl(lrun, b + r, 32);

  float* lo = ldsO + head * (16 * HIDDEN);
#pragma unroll
  for (int r = 0; r < 8; ++r) {
    const int row = r + 8 * half_id;
    lo[row * HIDDEN + 0 * 16 + m16] = acc0[r] * linv[r];
    lo[row * HIDDEN + 1 * 16 + m16] = acc1[r] * linv[r];
    lo[row * HIDDEN + 2 * 16 + m16] = acc2[r] * linv[r];
    lo[row * HIDDEN + 3 * 16 + m16] = acc3[r] * linv[r];
  }
  __syncthreads();

  for (int idx = threadIdx.x; idx < 16 * HIDDEN; idx += 256) {
    float sum = 0.0f;
#pragma unroll
    for (int w = 0; w < HEADS; ++w) sum += ldsO[w * (16 * HIDDEN) + idx];
    const int mm = idx >> 6;
    const int f = idx & 63;
    out[(size_t)(i0 + mm) * HIDDEN + f] = sum;
  }
}

// ---------------------------------------------------------------------------
extern "C" void kernel_launch(void* const* d_in, const int* in_sizes, int n_in,
                              void* d_out, int out_size, void* d_ws, size_t ws_size,
                              hipStream_t stream) {
  const float* x           = (const float*)d_in[0];
  const unsigned char* adj = (const unsigned char*)d_in[1];   // bool, 1 byte/elem
  const float* W           = (const float*)d_in[2];
  const float* attn_src    = (const float*)d_in[3];
  const float* attn_dst    = (const float*)d_in[4];
  float* out               = (float*)d_out;

  // Workspace carve-up (total ~12.25 MB)
  char* ws = (char*)d_ws;
  float*    h    = (float*)(ws);                              // 8 MB
  _Float16* hT16 = (_Float16*)(ws + (size_t)8 * 1024 * 1024); // 4 MB
  float*    srcT = (float*)(ws + (size_t)12 * 1024 * 1024);   // 128 KB
  float*    dstT = srcT + HEADS * N_NODES;                    // 128 KB

  // 1) h = x @ W^T  (f32 WMMA), also h^T in f16
  gemm_h_kernel<<<1024, 256, 0, stream>>>(x, W, h, hT16);
  // 2) attention scores, transposed layout
  scores_kernel<<<(N_NODES * HEADS) / 256, 256, 0, stream>>>(h, attn_src, attn_dst, srcT, dstT);
  // 3) fused masked softmax + alpha @ h + head mean (f16 WMMA, f32 accum)
  gat_attn_kernel<<<N_NODES / 16, 256, 0, stream>>>(adj, srcT, dstT, hT16, out);
}
